// ForwardDecoder_90340342104745
// MI455X (gfx1250) — compile-verified, matrix operand order checked by reading, not run on hardware
//
#include <hip/hip_runtime.h>
#include <hip/hip_bf16.h>
#include <math.h>

// ---------------------------------------------------------------------------
// Types for CDNA5 WMMA bf16 path
// ---------------------------------------------------------------------------
typedef __attribute__((ext_vector_type(16))) __bf16          v16bf;
typedef __attribute__((ext_vector_type(16))) unsigned short  v16u;
typedef __attribute__((ext_vector_type(8)))  unsigned short  v8u;
typedef __attribute__((ext_vector_type(8)))  float           v8f;

#define BQ   64
#define TQ   512
#define HID  1024
#define ENC  1024
#define EMBD 512
#define VV   32000
#define INW  2560     // EMB + ENC + HID
#define G4   4096     // 4*HID

__device__ __forceinline__ unsigned short f2bf(float f) {
    return __builtin_bit_cast(unsigned short, (__bf16)f);   // hw cvt, RNE
}

__device__ __forceinline__ float sigmoidf_(float x) {
    return 1.0f / (1.0f + __expf(-x));
}

// Build a 16-element bf16 WMMA operand fragment from two 8-bf16 chunks.
// ISA 16-bit operand layout: lane half h holds K = h*8.. and K = 16+h*8..
__device__ __forceinline__ v16bf frag16(const unsigned short* p0,
                                        const unsigned short* p1) {
    v8u a = *(const v8u*)p0;          // 16B -> global/ds load_b128
    v8u b = *(const v8u*)p1;
    v16u t;
#pragma unroll
    for (int j = 0; j < 8; ++j) { t[j] = a[j]; t[8 + j] = b[j]; }
    return __builtin_bit_cast(v16bf, t);
}

// ---------------------------------------------------------------------------
// One-time fp32 -> bf16 pack (n % 4 == 0 for all uses)
// ---------------------------------------------------------------------------
__global__ void cvt_f32_to_bf16(const float* __restrict__ in,
                                unsigned short* __restrict__ out, long n) {
    long i = ((long)blockIdx.x * blockDim.x + threadIdx.x) * 4;
    if (i >= n) return;
    float4 v = *(const float4*)(in + i);
    ushort4 o;
    o.x = f2bf(v.x); o.y = f2bf(v.y); o.z = f2bf(v.z); o.w = f2bf(v.w);
    *(ushort4*)(out + i) = o;
}

// ---------------------------------------------------------------------------
// WMMA GEMM on pre-packed bf16:  C[M,N] (=|+=) A[M,K] * Bt[N,K]^T (+bias[N])
// One wave per 16x16 tile, 8 waves/block. Inner loop: 4x b128 loads + wmma.
// M%16==0, N%16==0, K%64==0 for every call below.
// ---------------------------------------------------------------------------
__global__ void gemm_bt_wmma(const unsigned short* __restrict__ A, int lda,
                             const unsigned short* __restrict__ Bt, int ldb,
                             float* __restrict__ C, int ldc,
                             const float* __restrict__ bias,
                             int M, int N, int K, int accumulate) {
    const int ntiles = N >> 4;
    const int tiles  = (M >> 4) * ntiles;
    const int tileId = blockIdx.x * (blockDim.x >> 5) + (threadIdx.x >> 5);
    if (tileId >= tiles) return;               // wave-uniform exit

    const int mt   = tileId / ntiles;
    const int nt   = tileId % ntiles;
    const int lane = threadIdx.x & 31;
    const int half = lane >> 4;
    const int r    = lane & 15;

    const unsigned short* aRow = A  + (size_t)(mt * 16 + r) * lda + half * 8;
    const unsigned short* bRow = Bt + (size_t)(nt * 16 + r) * ldb + half * 8;

    v8f acc = {0.f, 0.f, 0.f, 0.f, 0.f, 0.f, 0.f, 0.f};
#pragma unroll 2
    for (int k0 = 0; k0 < K; k0 += 32) {
        v16bf a = frag16(aRow + k0, aRow + k0 + 16);
        v16bf b = frag16(bRow + k0, bRow + k0 + 16);
        acc = __builtin_amdgcn_wmma_f32_16x16x32_bf16(
                  false, a, false, b, (short)0, acc, false, false);
    }

    const int   n  = nt * 16 + r;
    const float bv = bias ? bias[n] : 0.f;
#pragma unroll
    for (int i = 0; i < 8; ++i) {
        const int    m   = mt * 16 + 8 * half + i;   // C layout: M = i + 8*half
        const size_t idx = (size_t)m * ldc + n;
        float val = acc[i] + bv;
        if (accumulate) val += C[idx];
        C[idx] = val;
    }
}

// ---------------------------------------------------------------------------
// Fused Bahdanau scores: e[b,t] = v . tanh(q[b] + Wk * keys[b,t])
// Keys tile (16 x 1024 fp32) is converted to bf16 into LDS ONCE per block;
// 8 waves then sweep the 64 N-tiles reading A-fragments with ds_load_b128
// and Wk (pre-packed bf16) with global b128 loads. k never hits memory.
// ---------------------------------------------------------------------------
__global__ void attn_scores_wmma(const float* __restrict__ keys,        // [B,T,D] fp32
                                 const unsigned short* __restrict__ Wk, // [1024,D] bf16
                                 const float* __restrict__ q,           // [B,1024]
                                 const float* __restrict__ vvec,        // [1024]
                                 const int*   __restrict__ mask,        // [B,T] or null
                                 float* __restrict__ e_out,             // [B,T]
                                 int T, int D) {
    __shared__ unsigned short smemA[16 * 1024];     // 32 KB of 320 KB WGP LDS
    __shared__ float e_smem[16];
    const int b  = blockIdx.x / (T >> 4);
    const int tt = blockIdx.x % (T >> 4);
    const int tid = threadIdx.x;
    if (tid < 16) e_smem[tid] = 0.f;

    // cooperative fill: thread -> (row = tid/16, 64-col segment = tid%16)
    {
        const int row  = tid >> 4;
        const int col0 = (tid & 15) * (D >> 4);     // D==1024 -> 64 cols
        const float* src = keys + ((size_t)b * T + tt * 16 + row) * D + col0;
        unsigned int* dst = (unsigned int*)(smemA + row * D + col0);
#pragma unroll
        for (int j = 0; j < 32; ++j) {              // 2 elems per packed store
            float2 v = *(const float2*)(src + 2 * j);
            dst[j] = (unsigned int)f2bf(v.x) | ((unsigned int)f2bf(v.y) << 16);
        }
    }
    __syncthreads();

    const int wave = tid >> 5;
    const int lane = tid & 31;
    const int half = lane >> 4;
    const int r    = lane & 15;

    const unsigned short* aBase = smemA + r * D + half * 8;
    const float* qbase = q + (size_t)b * 1024;

    float s[8];
#pragma unroll
    for (int i = 0; i < 8; ++i) s[i] = 0.f;

    for (int nt = wave; nt < 64; nt += 8) {          // 64 N-tiles of A=1024
        const unsigned short* bRow = Wk + (size_t)(nt * 16 + r) * D + half * 8;
        v8f acc = {0.f, 0.f, 0.f, 0.f, 0.f, 0.f, 0.f, 0.f};
#pragma unroll 2
        for (int k0 = 0; k0 < D; k0 += 32) {
            v16bf a  = frag16(aBase + k0, aBase + k0 + 16);   // ds_load_b128
            v16bf bm = frag16(bRow + k0, bRow + k0 + 16);     // global b128
            acc = __builtin_amdgcn_wmma_f32_16x16x32_bf16(
                      false, a, false, bm, (short)0, acc, false, false);
        }
        const int   n  = nt * 16 + r;     // fixed attention index per lane
        const float vn = vvec[n];
        const float qn = qbase[n];
#pragma unroll
        for (int i = 0; i < 8; ++i)       // t_local = i + 8*half, same all nt
            s[i] += tanhf(qn + acc[i]) * vn;
    }
#pragma unroll
    for (int i = 0; i < 8; ++i)
        atomicAdd(&e_smem[8 * half + i], s[i]);      // ds_add_f32 reduce
    __syncthreads();

    if (tid < 16) {
        const int t = tt * 16 + tid;
        float val = e_smem[tid];
        if (mask && mask[(size_t)b * T + t] == 0) val = -3.4028235e38f;
        e_out[(size_t)b * T + t] = val;
    }
}

// ---------------------------------------------------------------------------
// Softmax over T + ctx = attn @ keys. Streams keys once (memory-bound leg).
// ---------------------------------------------------------------------------
__global__ void softmax_ctx(const float* __restrict__ e,     // [B,T]
                            const float* __restrict__ keys,  // [B,T,D]
                            int T, int D,
                            float* __restrict__ ctx_out,     // x + col offset
                            int ctx_stride,
                            float* __restrict__ attn_out) {  // [B,T] or null
    __shared__ float red[256];
    __shared__ float attn_s[TQ];
    const int nchunk = D >> 8;
    const int b      = blockIdx.x / nchunk;
    const int d0     = (blockIdx.x % nchunk) << 8;
    const int tid    = threadIdx.x;
    const float* eb  = e + (size_t)b * T;

    float m = -3.4028235e38f;
    for (int t = tid; t < T; t += 256) m = fmaxf(m, eb[t]);
    red[tid] = m; __syncthreads();
    for (int s = 128; s > 0; s >>= 1) {
        if (tid < s) red[tid] = fmaxf(red[tid], red[tid + s]);
        __syncthreads();
    }
    const float maxv = red[0]; __syncthreads();

    float acc = 0.f;
    for (int t = tid; t < T; t += 256) acc += __expf(eb[t] - maxv);
    red[tid] = acc; __syncthreads();
    for (int s = 128; s > 0; s >>= 1) {
        if (tid < s) red[tid] += red[tid + s];
        __syncthreads();
    }
    const float inv = 1.0f / red[0]; __syncthreads();

    for (int t = tid; t < T; t += 256) {
        const float a = __expf(eb[t] - maxv) * inv;
        attn_s[t] = a;
        if (attn_out && d0 == 0) attn_out[(size_t)b * T + t] = a;
    }
    __syncthreads();

    const int d = d0 + tid;
    float cx = 0.f;
    const float* kb = keys + ((size_t)b * T) * D + d;
    for (int t = 0; t < T; ++t) cx += attn_s[t] * kb[(size_t)t * D];
    ctx_out[(size_t)b * ctx_stride + d] = cx;
}

// x[:, :EMB] = emb_table[y]
__global__ void embed_kernel(const int* __restrict__ y,
                             const float* __restrict__ tab,
                             float* __restrict__ x) {
    const int idx = blockIdx.x * blockDim.x + threadIdx.x;
    if (idx >= BQ * EMBD) return;
    const int b = idx / EMBD, e0 = idx % EMBD;
    x[(size_t)b * INW + e0] = tab[(size_t)y[b] * EMBD + e0];
}

// LSTM cell elementwise (torch gate order i,f,g,o)
__global__ void lstm_elem(const float* __restrict__ gates,
                          const float* __restrict__ c_old,
                          float* __restrict__ h_out,
                          float* __restrict__ c_out) {
    const int idx = blockIdx.x * blockDim.x + threadIdx.x;
    if (idx >= BQ * HID) return;
    const int b = idx / HID, j = idx % HID;
    const float* g = gates + (size_t)b * G4;
    const float ig = sigmoidf_(g[j]);
    const float fg = sigmoidf_(g[HID + j]);
    const float gg = tanhf(g[2 * HID + j]);
    const float og = sigmoidf_(g[3 * HID + j]);
    const float cn = fg * c_old[idx] + ig * gg;
    c_out[idx] = cn;
    h_out[idx] = og * tanhf(cn);
}

// p_gen[b] = sigmoid(c_enc.wc + h_new.ws + emb.wy + pg_b)
__global__ void pgen_kernel(const float* __restrict__ x,      // [B,INW]
                            const float* __restrict__ h_new,  // [B,HID]
                            const float* __restrict__ wc,
                            const float* __restrict__ wsv,
                            const float* __restrict__ wy,
                            const float* __restrict__ pg_b,
                            float* __restrict__ p_gen) {
    __shared__ float red[256];
    const int b = blockIdx.x, tid = threadIdx.x;
    const float* xb = x + (size_t)b * INW;
    float s = 0.f;
    for (int j = tid; j < ENC; j += 256)
        s += xb[EMBD + j] * wc[j] + h_new[(size_t)b * HID + j] * wsv[j];
    for (int j = tid; j < EMBD; j += 256) s += xb[j] * wy[j];
    red[tid] = s; __syncthreads();
    for (int t = 128; t > 0; t >>= 1) {
        if (tid < t) red[tid] += red[tid + t];
        __syncthreads();
    }
    if (tid == 0) p_gen[b] = sigmoidf_(red[0] + pg_b[0]);
}

// out[b,v] = p_gen[b] * softmax(logits[b,:])[v]
__global__ void vocab_final(const float* __restrict__ logits,
                            const float* __restrict__ p_gen,
                            float* __restrict__ out) {
    __shared__ float red[256];
    const int b = blockIdx.x, tid = threadIdx.x;
    const float* lb = logits + (size_t)b * VV;
    float m = -3.4028235e38f;
    for (int v = tid; v < VV; v += 256) m = fmaxf(m, lb[v]);
    red[tid] = m; __syncthreads();
    for (int s = 128; s > 0; s >>= 1) {
        if (tid < s) red[tid] = fmaxf(red[tid], red[tid + s]);
        __syncthreads();
    }
    const float maxv = red[0]; __syncthreads();
    float acc = 0.f;
    for (int v = tid; v < VV; v += 256) acc += __expf(lb[v] - maxv);
    red[tid] = acc; __syncthreads();
    for (int s = 128; s > 0; s >>= 1) {
        if (tid < s) red[tid] += red[tid + s];
        __syncthreads();
    }
    const float scale = p_gen[b] / red[0];
    for (int v = tid; v < VV; v += 256)
        out[(size_t)b * VV + v] = __expf(lb[v] - maxv) * scale;
}

// final_dist[b, src_ids[b,t]] += (1-p_gen[b]) * attn_enc[b,t]
__global__ void scatter_copy(const int* __restrict__ src_ids,
                             const float* __restrict__ attn_enc,
                             const float* __restrict__ p_gen,
                             float* __restrict__ out) {
    const int idx = blockIdx.x * blockDim.x + threadIdx.x;
    if (idx >= BQ * TQ) return;
    const int b  = idx / TQ;
    const int id = src_ids[idx];
    if (id < VV) {
        const float val = (1.0f - p_gen[b]) * attn_enc[idx];
        const int safe  = id < 0 ? 0 : id;
        atomicAdd(out + (size_t)b * VV + safe, val);
    }
}

// ---------------------------------------------------------------------------
extern "C" void kernel_launch(void* const* d_in, const int* in_sizes, int n_in,
                              void* d_out, int out_size, void* d_ws, size_t ws_size,
                              hipStream_t stream) {
    const int*   y        = (const int*)  d_in[0];
    const float* h        = (const float*)d_in[1];
    const float* c        = (const float*)d_in[2];
    const float* enc_out  = (const float*)d_in[3];
    const float* bwd      = (const float*)d_in[4];
    const int*   src_ids  = (const int*)  d_in[5];
    const int*   src_mask = (const int*)  d_in[6];
    const float* emb_tab  = (const float*)d_in[7];
    const float* W_ih     = (const float*)d_in[8];
    const float* W_hh     = (const float*)d_in[9];
    const float* b_ih     = (const float*)d_in[10];
    const float* b_hh     = (const float*)d_in[11];
    const float* Wq_e     = (const float*)d_in[12];
    const float* Wk_e     = (const float*)d_in[13];
    const float* v_e      = (const float*)d_in[14];
    const float* Wq_b     = (const float*)d_in[15];
    const float* Wk_b     = (const float*)d_in[16];
    const float* v_b      = (const float*)d_in[17];
    const float* Wp       = (const float*)d_in[18];
    const float* bp       = (const float*)d_in[19];
    const float* wc       = (const float*)d_in[20];
    const float* wsv      = (const float*)d_in[21];
    const float* wy       = (const float*)d_in[22];
    const float* pg_b     = (const float*)d_in[23];

    // d_out: final_dist [B,V] | h_new [B,HID] | c_new [B,HID]
    float* out_dist = (float*)d_out;
    float* h_new    = out_dist + (size_t)BQ * VV;
    float* c_new    = h_new + (size_t)BQ * HID;

    // ---- workspace: fp32 region (~10.8 MB) then bf16 region (~99 MB) ----
    float* w = (float*)d_ws;
    float* q_e      = w;
    float* q_b      = q_e      + BQ * HID;
    float* e_enc    = q_b      + BQ * HID;
    float* e_bwd    = e_enc    + BQ * TQ;
    float* attn_enc = e_bwd    + BQ * TQ;
    float* x        = attn_enc + BQ * TQ;
    float* gates    = x        + BQ * INW;
    float* logits   = gates    + BQ * G4;
    float* p_gen    = logits   + (size_t)BQ * VV;

    unsigned short* bf = (unsigned short*)(p_gen + 64);
    unsigned short* Wq_e_bf = bf;                                  // 1024x1024
    unsigned short* Wq_b_bf = Wq_e_bf + (size_t)HID * HID;
    unsigned short* Wk_e_bf = Wq_b_bf + (size_t)HID * HID;
    unsigned short* Wk_b_bf = Wk_e_bf + (size_t)HID * ENC;
    unsigned short* W_ih_bf = Wk_b_bf + (size_t)HID * HID;         // 4096x2560
    unsigned short* W_hh_bf = W_ih_bf + (size_t)G4 * INW;          // 4096x1024
    unsigned short* Wp_bf   = W_hh_bf + (size_t)G4 * HID;          // 32000x1024
    unsigned short* h_bf    = Wp_bf   + (size_t)VV * HID;
    unsigned short* x_bf    = h_bf    + (size_t)BQ * HID;
    unsigned short* hn_bf   = x_bf    + (size_t)BQ * INW;

    const dim3 blk(256);
    auto cvt_blocks  = [](long n) { return (int)((n / 4 + 255) / 256); };
    auto gemm_blocks = [](int M, int N) { return ((M / 16) * (N / 16) + 7) / 8; };

    // 0) one-time bf16 packing of all reused operands
    cvt_f32_to_bf16<<<cvt_blocks((long)HID * HID), blk, 0, stream>>>(Wq_e, Wq_e_bf, (long)HID * HID);
    cvt_f32_to_bf16<<<cvt_blocks((long)HID * HID), blk, 0, stream>>>(Wq_b, Wq_b_bf, (long)HID * HID);
    cvt_f32_to_bf16<<<cvt_blocks((long)HID * ENC), blk, 0, stream>>>(Wk_e, Wk_e_bf, (long)HID * ENC);
    cvt_f32_to_bf16<<<cvt_blocks((long)HID * HID), blk, 0, stream>>>(Wk_b, Wk_b_bf, (long)HID * HID);
    cvt_f32_to_bf16<<<cvt_blocks((long)G4 * INW), blk, 0, stream>>>(W_ih, W_ih_bf, (long)G4 * INW);
    cvt_f32_to_bf16<<<cvt_blocks((long)G4 * HID), blk, 0, stream>>>(W_hh, W_hh_bf, (long)G4 * HID);
    cvt_f32_to_bf16<<<cvt_blocks((long)VV * HID), blk, 0, stream>>>(Wp, Wp_bf, (long)VV * HID);
    cvt_f32_to_bf16<<<cvt_blocks((long)BQ * HID), blk, 0, stream>>>(h, h_bf, (long)BQ * HID);

    // 1) query projections  q = h @ Wq^T
    gemm_bt_wmma<<<gemm_blocks(BQ, HID), blk, 0, stream>>>(
        h_bf, HID, Wq_e_bf, HID, q_e, HID, nullptr, BQ, HID, HID, 0);
    gemm_bt_wmma<<<gemm_blocks(BQ, HID), blk, 0, stream>>>(
        h_bf, HID, Wq_b_bf, HID, q_b, HID, nullptr, BQ, HID, HID, 0);

    // 2) fused attention scores (34.4 GMAC each; k stays in VGPRs)
    attn_scores_wmma<<<BQ * (TQ / 16), blk, 0, stream>>>(
        enc_out, Wk_e_bf, q_e, v_e, src_mask, e_enc, TQ, ENC);
    attn_scores_wmma<<<BQ * (TQ / 16), blk, 0, stream>>>(
        bwd, Wk_b_bf, q_b, v_b, nullptr, e_bwd, TQ, HID);

    // 3) x = [emb | c_enc | c_bwd]
    embed_kernel<<<(BQ * EMBD + 255) / 256, blk, 0, stream>>>(y, emb_tab, x);
    softmax_ctx<<<BQ * (ENC / 256), blk, 0, stream>>>(
        e_enc, enc_out, TQ, ENC, x + EMBD, INW, attn_enc);
    softmax_ctx<<<BQ * (HID / 256), blk, 0, stream>>>(
        e_bwd, bwd, TQ, HID, x + EMBD + ENC, INW, nullptr);
    cvt_f32_to_bf16<<<cvt_blocks((long)BQ * INW), blk, 0, stream>>>(x, x_bf, (long)BQ * INW);

    // 4) gates = x @ W_ih^T + b_ih ;  gates += h @ W_hh^T + b_hh
    gemm_bt_wmma<<<gemm_blocks(BQ, G4), blk, 0, stream>>>(
        x_bf, INW, W_ih_bf, INW, gates, G4, b_ih, BQ, G4, INW, 0);
    gemm_bt_wmma<<<gemm_blocks(BQ, G4), blk, 0, stream>>>(
        h_bf, HID, W_hh_bf, HID, gates, G4, b_hh, BQ, G4, HID, 1);

    // 5) LSTM cell -> h_new, c_new (directly into d_out)
    lstm_elem<<<(BQ * HID + 255) / 256, blk, 0, stream>>>(gates, c, h_new, c_new);
    cvt_f32_to_bf16<<<cvt_blocks((long)BQ * HID), blk, 0, stream>>>(h_new, hn_bf, (long)BQ * HID);

    // 6) p_gen
    pgen_kernel<<<BQ, blk, 0, stream>>>(x, h_new, wc, wsv, wy, pg_b, p_gen);

    // 7) logits = h_new @ Wp^T + bp
    gemm_bt_wmma<<<gemm_blocks(BQ, VV), blk, 0, stream>>>(
        hn_bf, HID, Wp_bf, HID, logits, VV, bp, BQ, VV, HID, 0);

    // 8) final_dist = p_gen*softmax(logits) ; then pointer-copy scatter
    vocab_final<<<BQ, blk, 0, stream>>>(logits, p_gen, out_dist);
    scatter_copy<<<(BQ * TQ + 255) / 256, blk, 0, stream>>>(
        src_ids, attn_enc, p_gen, out_dist);
}
